// PrototypeContrastiveLoss_7636451852444
// MI455X (gfx1250) — matrix operand order repaired and checked by reading, not compiled, not run on hardware
//
#include <hip/hip_runtime.h>
#include <hip/hip_bf16.h>

// ---------------------------------------------------------------------------
// PrototypeContrastiveLoss on MI455X (gfx1250).
// p2p term = B*N*N*C*2 = 17.2 GFLOP over ~16MB -> compute-bound -> f16 WMMA.
// Each wave owns a 32-row stripe (two 16x16 WMMA chains sharing each B tile).
// B tiles are explicitly double-buffered so next-tile loads overlap the
// current tile's 8 WMMAs and fill the WMMA->VALU hazard slots. The exp scale
// 1/(tau*ln2) is pre-baked into the A-side operand, so the epilogue is
// exactly 1 v_exp_f32 + 2 FMA per matrix element.
// ---------------------------------------------------------------------------

typedef _Float16 half8 __attribute__((ext_vector_type(8)));
typedef _Float16 v16h  __attribute__((ext_vector_type(16)));
typedef float    v8f   __attribute__((ext_vector_type(8)));

#define KCLS      4
#define IGN       4
#define BSZ       4
#define CCH       128
#define NPX       4096          // 64*64
#define HL        256
#define WL        256
#define EXP2_TAU  3.60673760222f   // (1/0.4) / ln(2)
#define EPS_L     1e-8f

__device__ __forceinline__ float waveRedSum(float v) {
    v += __shfl_xor(v, 16, 32);
    v += __shfl_xor(v,  8, 32);
    v += __shfl_xor(v,  4, 32);
    v += __shfl_xor(v,  2, 32);
    v += __shfl_xor(v,  1, 32);
    return v;
}

// scal[0]=p2p_sum  scal[1]=p2proto_sumA  scal[2]=p2proto_sumB
// scal[3]=validCntA  scal[4]=validCntB
__global__ void init_scalars_k(float* s) {
    if (threadIdx.x < 8) s[threadIdx.x] = 0.f;
}

// Nearest interp (src = floor(dst*256/64) = dst*4), label caching + valid counts.
__global__ void make_labels_k(const int* __restrict__ label,
                              const int* __restrict__ label_aug,
                              int* __restrict__ labA, int* __restrict__ labB,
                              float* scal) {
    int idx = blockIdx.x * blockDim.x + threadIdx.x;   // 0..B*N-1
    int b = idx / NPX, n = idx % NPX;
    int h = n >> 6, w = n & 63;
    int src = b * (HL * WL) + (h * 4) * WL + (w * 4);
    int la = label[src], lb = label_aug[src];
    labA[idx] = la;
    labB[idx] = lb;
    float va = (la != IGN) ? 1.f : 0.f;
    float vb = (lb != IGN) ? 1.f : 0.f;
    va = waveRedSum(va);
    vb = waveRedSum(vb);
    if ((threadIdx.x & 31) == 0) {
        atomicAdd(&scal[3], va);
        atomicAdd(&scal[4], vb);
    }
}

// L2-normalize over C and transpose [b][c][n] f32 -> [b][n][c] f16.
// Set A (grid.y==0) is additionally pre-scaled by EXP2_TAU so the p2p and
// p2proto epilogues can use exp2() with no extra multiply.
// 4 threads per pixel, 32 channels each; 64 pixels per 256-thread block.
__global__ void __launch_bounds__(256)
normalize_f_k(const float* __restrict__ f, const float* __restrict__ f_aug,
              _Float16* __restrict__ outA, _Float16* __restrict__ outB) {
    const float* src = blockIdx.y ? f_aug : f;
    _Float16*    dst = blockIdx.y ? outB  : outA;
    float scale      = blockIdx.y ? 1.f   : EXP2_TAU;
    int t = threadIdx.x;
    int P = blockIdx.x * 64 + (t >> 2);   // global pixel = b*N + n
    int q = t & 3;
    int b = P / NPX, n = P % NPX;
    const float* base = src + (size_t)b * CCH * NPX + n;
    float v[32];
    float ss = 0.f;
#pragma unroll
    for (int i = 0; i < 32; ++i) {
        float x = base[(size_t)(q * 32 + i) * NPX];
        v[i] = x;
        ss = fmaf(x, x, ss);
    }
    ss += __shfl_xor(ss, 1, 32);
    ss += __shfl_xor(ss, 2, 32);
    float inv = scale / fmaxf(sqrtf(ss), 1e-12f);
    _Float16* o = dst + (size_t)P * CCH + q * 32;
#pragma unroll
    for (int i = 0; i < 32; ++i) o[i] = (_Float16)(v[i] * inv);
}

// Normalize 2K=8 prototypes -> f32 [8][128]; one wave per prototype.
__global__ void __launch_bounds__(256)
normalize_proto_k(const float* __restrict__ proto,
                  const float* __restrict__ proto_aug,
                  float* __restrict__ protoNorm) {
    int t = threadIdx.x;
    int p = t >> 5, l = t & 31;
    const float* src = (p < KCLS) ? (proto + p * CCH) : (proto_aug + (p - KCLS) * CCH);
    float v[4];
    float ss = 0.f;
#pragma unroll
    for (int i = 0; i < 4; ++i) {
        float x = src[l + 32 * i];
        v[i] = x;
        ss = fmaf(x, x, ss);
    }
    ss = waveRedSum(ss);
    float inv = 1.f / fmaxf(sqrtf(ss), 1e-12f);
#pragma unroll
    for (int i = 0; i < 4; ++i) protoNorm[p * CCH + l + 32 * i] = v[i] * inv;
}

// ---------------------------------------------------------------------------
// p2p main kernel (see header comment).
// ---------------------------------------------------------------------------
__global__ void __launch_bounds__(128)
p2p_kernel(const _Float16* __restrict__ fnA, const _Float16* __restrict__ fnB,
           const int* __restrict__ labA, const int* __restrict__ labB,
           float* scal) {
    int wave = threadIdx.x >> 5;
    int lane = threadIdx.x & 31;
    int lo = lane & 15, hi = lane >> 4;
    int gw = blockIdx.x * 4 + wave;          // 0 .. B*(N/32)-1 = 511
    int b = gw >> 7;                          // 128 stripes per batch
    int rowBase = (gw & 127) << 5;            // 32-row stripe
    int bN = b * NPX;

    // Two A fragments (16x32 f16 per chunk):
    // halves h<8 -> K = hi*8+h ; h>=8 -> K = 16+hi*8+(h-8)
    v16h a0[4], a1[4];
#pragma unroll
    for (int s = 0; s < 2; ++s) {
        const _Float16* arow = fnA + (size_t)(bN + rowBase + s * 16 + lo) * CCH;
#pragma unroll
        for (int kc = 0; kc < 4; ++kc) {
            const half8* p0 = (const half8*)(arow + kc * 32 + hi * 8);
            half8 l8 = p0[0];
            half8 h8 = p0[2];                // +16 halves
            v16h av = __builtin_shufflevector(l8, h8, 0, 1, 2, 3, 4, 5, 6, 7,
                                              8, 9, 10, 11, 12, 13, 14, 15);
            if (s == 0) a0[kc] = av; else a1[kc] = av;
        }
    }

    // Row labels of this wave-half's D rows (D layout: lanes 0-15 -> M=r,
    // lanes 16-31 -> M=r+8). Same for all 16 lanes of a half.
    int rl0[8], rl1[8];
#pragma unroll
    for (int r = 0; r < 8; ++r) {
        rl0[r] = labA[bN + rowBase +      r + 8 * hi];
        rl1[r] = labA[bN + rowBase + 16 + r + 8 * hi];
    }

    float p0a[8], v0a[8], p1a[8], v1a[8];
#pragma unroll
    for (int r = 0; r < 8; ++r) { p0a[r] = 0.f; v0a[r] = 0.f; p1a[r] = 0.f; v1a[r] = 0.f; }

    const _Float16* bbase = fnB + (size_t)bN * CCH;

    // B tile (32x16): halves h<8 -> K = hi*8+h ; h>=8 -> K = 16+hi*8+(h-8).
    auto loadB = [&](int jt, v16h bm[4]) {
        const _Float16* brow = bbase + (size_t)(jt + lo) * CCH;
        __builtin_prefetch(brow + 32 * CCH, 0, 1);   // 2 tiles ahead
#pragma unroll
        for (int kc = 0; kc < 4; ++kc) {
            const half8* p0 = (const half8*)(brow + kc * 32 + hi * 8);
            half8 l8 = p0[0];
            half8 h8 = p0[2];
            bm[kc] = __builtin_shufflevector(l8, h8, 0, 1, 2, 3, 4, 5, 6, 7,
                                             8, 9, 10, 11, 12, 13, 14, 15);
        }
    };

    // 8 WMMAs on a staged B tile + exp/accumulate epilogue.
    auto process = [&](int jt, const v16h bm[4]) {
        v8f d0 = {}, d1 = {};
#pragma unroll
        for (int kc = 0; kc < 4; ++kc) {
            d0 = __builtin_amdgcn_wmma_f32_16x16x32_f16(
                     false, a0[kc], false, bm[kc], (short)0, d0, false, false);
            d1 = __builtin_amdgcn_wmma_f32_16x16x32_f16(
                     false, a1[kc], false, bm[kc], (short)0, d1, false, false);
        }
        int cls = labB[bN + jt + lo];              // lane-uniform column class
        float vm = (cls != IGN) ? 1.f : 0.f;       // valid-column mask
        float pm0[8], pm1[8];
#pragma unroll
        for (int r = 0; r < 8; ++r) {              // positive masks per row
            pm0[r] = (cls == rl0[r]) ? 1.f : 0.f;
            pm1[r] = (cls == rl1[r]) ? 1.f : 0.f;
        }
#pragma unroll
        for (int r = 0; r < 8; ++r) {
            float e0 = __builtin_amdgcn_exp2f(d0[r]);   // scale baked into fnA
            float e1 = __builtin_amdgcn_exp2f(d1[r]);
            p0a[r] = fmaf(e0, pm0[r], p0a[r]);
            v0a[r] = fmaf(e0, vm,     v0a[r]);
            p1a[r] = fmaf(e1, pm1[r], p1a[r]);
            v1a[r] = fmaf(e1, vm,     v1a[r]);
        }
    };

    // Software-pipelined double buffer: next tile's loads overlap this tile's
    // WMMAs (tail load redirected to tile 0; in-bounds, result unused).
    v16h b0[4], b1[4];
    loadB(0, b0);
    for (int jt = 0; jt < NPX; jt += 32) {
        loadB(jt + 16, b1);
        process(jt, b0);
        loadB((jt + 32 < NPX) ? jt + 32 : 0, b0);
        process(jt + 16, b1);
    }

    // Reduce over the 16 lanes of each half (lanes 0-15: rows r, 16-31: rows r+8)
#pragma unroll
    for (int r = 0; r < 8; ++r) {
#pragma unroll
        for (int m = 1; m <= 8; m <<= 1) {
            p0a[r] += __shfl_xor(p0a[r], m, 32);
            v0a[r] += __shfl_xor(v0a[r], m, 32);
            p1a[r] += __shfl_xor(p1a[r], m, 32);
            v1a[r] += __shfl_xor(v1a[r], m, 32);
        }
    }

    if (lo == 0) {
        float lsum = 0.f;
#pragma unroll
        for (int r = 0; r < 8; ++r) {
            float loss0 = -__logf((p0a[r] + EPS_L) / (v0a[r] + EPS_L));
            float loss1 = -__logf((p1a[r] + EPS_L) / (v1a[r] + EPS_L));
            lsum += (rl0[r] != IGN) ? loss0 : 0.f;
            lsum += (rl1[r] != IGN) ? loss1 : 0.f;
        }
        atomicAdd(&scal[0], lsum);
    }
}

// p2proto for both feature sets (grid.y selects); cheap: 67 MFLOP total.
// fnA carries the EXP2_TAU scale already; for the fnB term the prototypes
// are scaled instead.
__global__ void __launch_bounds__(256)
p2proto_kernel(const _Float16* __restrict__ fnA, const _Float16* __restrict__ fnB,
               const int* __restrict__ labA, const int* __restrict__ labB,
               const float* __restrict__ protoNorm, float* scal) {
    __shared__ float pr[8 * CCH];
    int set = blockIdx.y;
    float ps = set ? EXP2_TAU : 1.f;
    for (int i = threadIdx.x; i < 8 * CCH; i += blockDim.x) pr[i] = protoNorm[i] * ps;
    __syncthreads();

    const _Float16* fn = set ? fnB : fnA;
    const int* lab     = set ? labB : labA;
    int idx = blockIdx.x * blockDim.x + threadIdx.x;   // 0..B*N-1

    float dot[8];
#pragma unroll
    for (int p = 0; p < 8; ++p) dot[p] = 0.f;
    const _Float16* frow = fn + (size_t)idx * CCH;
    for (int c = 0; c < CCH; ++c) {
        float x = (float)frow[c];
#pragma unroll
        for (int p = 0; p < 8; ++p) dot[p] = fmaf(x, pr[p * CCH + c], dot[p]);
    }
    float e[8];
    float all = 0.f;
#pragma unroll
    for (int p = 0; p < 8; ++p) {
        e[p] = __builtin_amdgcn_exp2f(dot[p]);   // scale already applied
        all += e[p];
    }
    int l = lab[idx];
    int lv = (l == IGN) ? 0 : l;
    float pos = (lv == 0) ? (e[0] + e[4])
              : (lv == 1) ? (e[1] + e[5])
              : (lv == 2) ? (e[2] + e[6]) : (e[3] + e[7]);
    float loss = -__logf((pos + EPS_L) / (all + EPS_L));
    float contrib = (l != IGN) ? loss : 0.f;
    contrib = waveRedSum(contrib);
    if ((threadIdx.x & 31) == 0) atomicAdd(&scal[1 + set], contrib);
}

__global__ void finalize_k(const float* scal, float* out) {
    if (threadIdx.x == 0 && blockIdx.x == 0) {
        float cntA = fmaxf(scal[3], 1.f);
        float cntB = fmaxf(scal[4], 1.f);
        float p2p   = scal[0] / cntA;
        float p2pro = scal[1] / cntA + scal[2] / cntB;
        out[0] = 0.4f * p2p + 0.4f * p2pro;
    }
}

extern "C" void kernel_launch(void* const* d_in, const int* in_sizes, int n_in,
                              void* d_out, int out_size, void* d_ws, size_t ws_size,
                              hipStream_t stream) {
    (void)in_sizes; (void)n_in; (void)out_size; (void)ws_size;
    const float* f         = (const float*)d_in[0];
    const float* f_aug     = (const float*)d_in[1];
    const float* proto     = (const float*)d_in[2];
    const float* proto_aug = (const float*)d_in[3];
    const int*   label     = (const int*)d_in[4];
    const int*   label_aug = (const int*)d_in[5];
    float* out = (float*)d_out;

    // Workspace carving (~8.4 MB): normalized f16 features + labels + protos + scalars
    char* ws = (char*)d_ws;
    size_t o = 0;
    _Float16* fnA = (_Float16*)(ws + o); o += (size_t)BSZ * NPX * CCH * sizeof(_Float16);
    _Float16* fnB = (_Float16*)(ws + o); o += (size_t)BSZ * NPX * CCH * sizeof(_Float16);
    int* labA = (int*)(ws + o); o += (size_t)BSZ * NPX * sizeof(int);
    int* labB = (int*)(ws + o); o += (size_t)BSZ * NPX * sizeof(int);
    float* protoNorm = (float*)(ws + o); o += 8 * CCH * sizeof(float);
    float* scal = (float*)(ws + o); o += 64;

    init_scalars_k<<<1, 32, 0, stream>>>(scal);
    make_labels_k<<<(BSZ * NPX) / 256, 256, 0, stream>>>(label, label_aug, labA, labB, scal);
    dim3 gN((BSZ * NPX) / 64, 2);
    normalize_f_k<<<gN, 256, 0, stream>>>(f, f_aug, fnA, fnB);
    normalize_proto_k<<<1, 256, 0, stream>>>(proto, proto_aug, protoNorm);
    p2p_kernel<<<(BSZ * (NPX / 32)) / 4, 128, 0, stream>>>(fnA, fnB, labA, labB, scal);
    dim3 gP((BSZ * NPX) / 256, 2);
    p2proto_kernel<<<gP, 256, 0, stream>>>(fnA, fnB, labA, labB, protoNorm, scal);
    finalize_k<<<1, 32, 0, stream>>>(scal, out);
}
